// GPT2_69466801045848
// MI455X (gfx1250) — matrix operand order repaired and checked by reading, compiled, not run on hardware
//
#include <hip/hip_runtime.h>
#include <hip/hip_bf16.h>

// ---------------- constants ----------------
constexpr int Lc = 12;
constexpr int Dc = 768;
constexpr int Hc = 12;
constexpr int Tc = 1024;
constexpr int Vc = 50257;
constexpr int Bc = 8;
constexpr int BTc = Bc * Tc;    // 8192 rows
constexpr int HDc = Dc / Hc;    // 64

// ---------------- types ----------------
typedef __attribute__((ext_vector_type(16))) __bf16 v16bf;
typedef __attribute__((ext_vector_type(8)))  float  v8f;
typedef __attribute__((ext_vector_type(4)))  unsigned int u32x4;
typedef __attribute__((ext_vector_type(8)))  int i32x8;
typedef __attribute__((ext_vector_type(4)))  int i32x4;
typedef __attribute__((address_space(3))) unsigned short lds_us_t;

__device__ inline unsigned short f2bf(float x) {
    unsigned int u = __float_as_uint(x);
    unsigned int r = u + 0x7FFFu + ((u >> 16) & 1u);   // round-to-nearest-even
    return (unsigned short)(r >> 16);
}

__device__ inline v8f zero8() {
    v8f z;
#pragma unroll
    for (int i = 0; i < 8; ++i) z[i] = 0.0f;
    return z;
}

// A-matrix fragment (16x32 bf16 per K-step) from a row-major bf16 tile.
// Layout (ISA 7.12.2): lanes 0-15 -> M=lane, elems 0..7 -> K=half*8+e,
// elems 8..15 -> K=16+half*8+(e-8), where half = lane>>4.
__device__ inline v16bf load_a_frag(const unsigned short* base, int ld, int row,
                                    int kbase, int lane) {
    int half = lane >> 4;
    union { v16bf v; uint4 q[2]; } f;
    const unsigned short* p = base + (size_t)row * ld + kbase + half * 8;
    f.q[0] = *(const uint4*)p;
    f.q[1] = *(const uint4*)(p + 16);
    return f.v;
}

// B-matrix fragment (32x16 bf16 per K-step) from a COLUMN-MAJOR [N][K] LDS
// tile (K contiguous per column). Lane -> column N = col + (lane&15),
// elems e<8 -> K = kbase + (lane>>4)*8 + e, e>=8 -> K = kbase + 16 + (lane>>4)*8 + (e-8) => 2x b128.
__device__ inline v16bf load_bT_frag(const unsigned short* Bs, int ldK, int col,
                                     int kbase, int lane) {
    const unsigned short* p =
        Bs + (size_t)(col + (lane & 15)) * ldK + kbase + (lane >> 4) * 8;
    union { v16bf v; uint4 q[2]; } f;
    f.q[0] = *(const uint4*)p;
    f.q[1] = *(const uint4*)(p + 16);
    return f.v;
}

// ---------------- Tensor Data Mover: 2D bf16 tile -> LDS ----------------
#if __has_builtin(__builtin_amdgcn_tensor_load_to_lds)
#define HAVE_TDM 1
__device__ inline void tdm_load_2d_bf16(const void* gaddr, unsigned ldsOff,
                                        unsigned tensor_d0, unsigned tensor_d1,
                                        unsigned tile_d0, unsigned tile_d1,
                                        unsigned d0_stride /*elements*/) {
    unsigned long long ga = (unsigned long long)gaddr;
    u32x4 g0;
    g0[0] = 1u;                                   // count=1 valid descriptor
    g0[1] = ldsOff;                               // LDS byte address
    g0[2] = (unsigned)(ga & 0xFFFFFFFFu);         // global addr [31:0]
    g0[3] = (unsigned)((ga >> 32) & 0x01FFFFFFu)  // global addr [56:32]
            | (2u << 30);                         // type = 2 ("image")
    i32x8 g1;
    g1[0] = (int)(1u << 16);                      // data_size=1 -> 2 bytes
    g1[1] = (int)((tensor_d0 & 0xFFFFu) << 16);   // tensor_dim0 [15:0]
    g1[2] = (int)(((tensor_d0 >> 16) & 0xFFFFu) |
                  ((tensor_d1 & 0xFFFFu) << 16)); // dim0 hi | dim1 lo
    g1[3] = (int)(((tensor_d1 >> 16) & 0xFFFFu) |
                  ((tile_d0 & 0xFFFFu) << 16));   // dim1 hi | tile_dim0
    g1[4] = (int)(tile_d1 & 0xFFFFu);             // tile_dim1 (tile_dim2=0)
    g1[5] = (int)d0_stride;                       // tensor_dim0_stride lo32
    g1[6] = 0;
    g1[7] = 0;
    i32x4 zz4; zz4[0] = 0; zz4[1] = 0; zz4[2] = 0; zz4[3] = 0;
    i32x8 zz8;
#pragma unroll
    for (int i = 0; i < 8; ++i) zz8[i] = 0;
    // 6-arg form (clang-23 / therock HIP headers): two descriptor groups,
    // two optional groups, one extra group, cache policy.
    __builtin_amdgcn_tensor_load_to_lds(g0, g1, zz4, zz4, zz8, 0);
}
#else
#define HAVE_TDM 0
#endif

// ---------------- embedding ----------------
__global__ __launch_bounds__(256) void embed_kernel(
    const int* __restrict__ idx, const float* __restrict__ wte,
    const float* __restrict__ wpe, float* __restrict__ x) {
    int row = blockIdx.x;
    int t = row & (Tc - 1);
    int tok = idx[row];
    for (int i = threadIdx.x; i < Dc; i += 256)
        x[(size_t)row * Dc + i] = wte[(size_t)tok * Dc + i] + wpe[(size_t)t * Dc + i];
}

// ---------------- layernorm (f32 in -> bf16 out) ----------------
__device__ inline float block_sum256(float v, float* red, int tid) {
    int lane = tid & 31, wave = tid >> 5;
#pragma unroll
    for (int m = 16; m > 0; m >>= 1) v += __shfl_xor(v, m, 32);
    if (lane == 0) red[wave] = v;
    __syncthreads();
    float t = 0.0f;
#pragma unroll
    for (int w = 0; w < 8; ++w) t += red[w];
    __syncthreads();
    return t;
}

__global__ __launch_bounds__(256) void ln_kernel(
    const float* __restrict__ x, const float* __restrict__ w,
    const float* __restrict__ b, unsigned short* __restrict__ out,
    int rowStride, int rowOffset) {
    __shared__ float red[8];
    int r = blockIdx.x;
    int xr = r * rowStride + rowOffset;
    const float* xp = &x[(size_t)xr * Dc];
    int tid = threadIdx.x;

    float s = 0.0f;
    for (int i = tid; i < Dc; i += 256) s += xp[i];
    float mu = block_sum256(s, red, tid) * (1.0f / Dc);

    float vs = 0.0f;
    for (int i = tid; i < Dc; i += 256) { float d = xp[i] - mu; vs += d * d; }
    float var = block_sum256(vs, red, tid) * (1.0f / Dc);
    float rstd = rsqrtf(var + 1e-5f);

    for (int i = tid; i < Dc; i += 256)
        out[(size_t)r * Dc + i] = f2bf((xp[i] - mu) * rstd * w[i] + b[i]);
}

// ---------------- WMMA GEMM: C = A(bf16 MxK) * B(f32->bf16 KxN) ----------------
// EPI 0: out_bf16 = C + bias
// EPI 1: out_bf16 = gelu(C + bias)        (exact erf GELU)
// EPI 2: out_f32 += C + bias              (residual accumulate)
template <int EPI>
__global__ __launch_bounds__(256) void gemm_bf16(
    const unsigned short* __restrict__ A, const float* __restrict__ B,
    const float* __restrict__ bias, unsigned short* __restrict__ outBF,
    float* __restrict__ outF, int M, int N, int K) {
    __shared__ unsigned short As[64 * 32];     // [m][k] row-major
    __shared__ unsigned short Bs[128 * 32];    // [n][k] column-major tile
    int tid = threadIdx.x, lane = tid & 31, wave = tid >> 5;
    int wm = wave >> 1, wn = wave & 1;         // 4 x 2 wave grid
    int m0 = blockIdx.y * 64, n0 = blockIdx.x * 128;

    v8f acc[4];
#pragma unroll
    for (int n = 0; n < 4; ++n) acc[n] = zero8();

#if HAVE_TDM
    unsigned asOff = (unsigned)(size_t)(lds_us_t*)As;
#endif

    for (int k0 = 0; k0 < K; k0 += 32) {
#if HAVE_TDM
        if (wave == 0) {   // TDM: DMA the 64x32 bf16 A tile into LDS
            tdm_load_2d_bf16(&A[(size_t)m0 * K + k0], asOff,
                             (unsigned)K, (unsigned)M, 32u, 64u, (unsigned)K);
        }
#else
        {   // A tile: 64x32 bf16, 8 elems/thread, vectorized
            int r = tid >> 2, c = (tid & 3) * 8;
            *(uint4*)&As[r * 32 + c] =
                *(const uint4*)&A[(size_t)(m0 + r) * K + k0 + c];
        }
#endif
        {   // B tile: 32x128 f32 -> bf16, stored [n][k] in LDS
            int r = tid >> 3, c = (tid & 7) * 16;      // r = k row, c = n base
            const float* src = &B[(size_t)(k0 + r) * N + n0 + c];
            unsigned short tmp[16];
#pragma unroll
            for (int j = 0; j < 16; j += 4) {
                float4 f = *(const float4*)(src + j);
                tmp[j] = f2bf(f.x); tmp[j + 1] = f2bf(f.y);
                tmp[j + 2] = f2bf(f.z); tmp[j + 3] = f2bf(f.w);
            }
#pragma unroll
            for (int j = 0; j < 16; ++j) Bs[(c + j) * 32 + r] = tmp[j];
            if (k0 + 32 < K)   // gfx1250 global_prefetch of next B tile
                __builtin_prefetch(&B[(size_t)(k0 + 32 + r) * N + n0 + c], 0, 0);
        }
#if HAVE_TDM
        if (wave == 0) __builtin_amdgcn_s_wait_tensorcnt(0);
#endif
        __syncthreads();
        v16bf a = load_a_frag(As, 32, wm * 16 + (lane & 15), 0, lane);
#pragma unroll
        for (int n = 0; n < 4; ++n) {
            v16bf bfr = load_bT_frag(Bs, 32, wn * 64 + n * 16, 0, lane);
            acc[n] = __builtin_amdgcn_wmma_f32_16x16x32_bf16(
                false, a, false, bfr, (short)0, acc[n], false, false);
        }
        __syncthreads();
    }

    int nlane = lane & 15, rb = (lane >> 4) * 8;
#pragma unroll
    for (int n = 0; n < 4; ++n) {
        int col = n0 + wn * 64 + n * 16 + nlane;
        float bv = bias[col];
#pragma unroll
        for (int r = 0; r < 8; ++r) {
            int row = m0 + wm * 16 + rb + r;
            float v = acc[n][r] + bv;
            if (EPI == 0) {
                outBF[(size_t)row * N + col] = f2bf(v);
            } else if (EPI == 1) {
                float g = 0.5f * v * (1.0f + erff(v * 0.70710678118f));
                outBF[(size_t)row * N + col] = f2bf(g);
            } else {
                outF[(size_t)row * N + col] += v;
            }
        }
    }
}

// ---------------- flash attention (bf16 WMMA, f32 softmax) ----------------
__global__ __launch_bounds__(128) void attn_kernel(
    const unsigned short* __restrict__ qkv,   // [B*T, 3*D] bf16 (q|k|v)
    unsigned short* __restrict__ o) {          // [B*T, D] bf16
    __shared__ unsigned short Kt[64 * 64];     // [key][d]  (natural layout)
    __shared__ unsigned short Vt[64 * 64];     // [d][key]  (transposed)
    __shared__ unsigned short Ps[4][16 * 64];  // per-wave P [row][key]

    int qt = blockIdx.x, h = blockIdx.y, b = blockIdx.z;
    int tid = threadIdx.x, lane = tid & 31, wave = tid >> 5;
    int q0 = qt * 64;
    size_t rowbase = (size_t)b * Tc;
    const int QKVW = 3 * Dc;

    // Q fragments (held in registers for the whole KV loop)
    v16bf qf[2];
    {
        int m = lane & 15;
        const unsigned short* qp =
            &qkv[(rowbase + q0 + wave * 16 + m) * QKVW + h * HDc];
#pragma unroll
        for (int s = 0; s < 2; ++s) qf[s] = load_a_frag(qp, 0, 0, s * 32, lane);
    }

    float mrow[8], lrow[8];
    v8f oacc[4];
#pragma unroll
    for (int r = 0; r < 8; ++r) { mrow[r] = -3.0e38f; lrow[r] = 0.0f; }
#pragma unroll
    for (int n = 0; n < 4; ++n) oacc[n] = zero8();

    int nlane = lane & 15, rb = (lane >> 4) * 8;
    const float scale = 0.125f;   // 1/sqrt(64)

    for (int kv0 = 0; kv0 <= q0; kv0 += 64) {
        {   // load K (natural) and V (transposed) tiles: 2 threads per key row
            int key = tid >> 1, dbase = (tid & 1) * 32;
            const unsigned short* kp =
                &qkv[(rowbase + kv0 + key) * QKVW + Dc + h * HDc + dbase];
            const unsigned short* vp = kp + Dc;
            union { uint4 q[4]; unsigned short u[32]; } vv;
#pragma unroll
            for (int j = 0; j < 4; ++j) {   // K: b128 load -> b128 LDS store
                *(uint4*)&Kt[key * 64 + dbase + j * 8] = *(const uint4*)(kp + j * 8);
                vv.q[j] = *(const uint4*)(vp + j * 8);
            }
#pragma unroll
            for (int j = 0; j < 32; ++j) Vt[(dbase + j) * 64 + key] = vv.u[j];
        }
        __syncthreads();

        // S = Q * K^T  (16 queries x 64 keys per wave); B cols = keys, K-dim = d
        v8f sacc[4];
#pragma unroll
        for (int n = 0; n < 4; ++n) sacc[n] = zero8();
#pragma unroll
        for (int s = 0; s < 2; ++s) {
#pragma unroll
            for (int n = 0; n < 4; ++n) {
                v16bf kb = load_bT_frag(Kt, 64, n * 16, s * 32, lane);
                sacc[n] = __builtin_amdgcn_wmma_f32_16x16x32_bf16(
                    false, qf[s], false, kb, (short)0, sacc[n], false, false);
            }
        }

        // online softmax update
#pragma unroll
        for (int r = 0; r < 8; ++r) {
            int qrow = q0 + wave * 16 + rb + r;
            float sv[4];
            float lm = -3.0e38f;
#pragma unroll
            for (int n = 0; n < 4; ++n) {
                int key = kv0 + n * 16 + nlane;
                float v = sacc[n][r] * scale;
                if (key > qrow) v = -3.0e38f;
                sv[n] = v;
                lm = fmaxf(lm, v);
            }
#pragma unroll
            for (int m = 1; m <= 8; m <<= 1) lm = fmaxf(lm, __shfl_xor(lm, m, 32));
            float mnew = fmaxf(mrow[r], lm);
            float ps = 0.0f;
#pragma unroll
            for (int n = 0; n < 4; ++n) {
                float p = __expf(sv[n] - mnew);
                ps += p;
                Ps[wave][(rb + r) * 64 + n * 16 + nlane] = f2bf(p);
            }
#pragma unroll
            for (int m = 1; m <= 8; m <<= 1) ps += __shfl_xor(ps, m, 32);
            float alpha = __expf(mrow[r] - mnew);
            lrow[r] = lrow[r] * alpha + ps;
            mrow[r] = mnew;
#pragma unroll
            for (int n = 0; n < 4; ++n) oacc[n][r] *= alpha;
        }
        __syncthreads();   // Ps visible to all lanes of the wave

        // O += P * V ; B cols = d, K-dim = keys
#pragma unroll
        for (int s = 0; s < 2; ++s) {
            v16bf ap = load_a_frag(Ps[wave], 64, lane & 15, s * 32, lane);
#pragma unroll
            for (int n = 0; n < 4; ++n) {
                v16bf vb = load_bT_frag(Vt, 64, n * 16, s * 32, lane);
                oacc[n] = __builtin_amdgcn_wmma_f32_16x16x32_bf16(
                    false, ap, false, vb, (short)0, oacc[n], false, false);
            }
        }
        __syncthreads();   // before K/V tiles are overwritten
    }

    // write O (normalized) as bf16
#pragma unroll
    for (int n = 0; n < 4; ++n) {
#pragma unroll
        for (int r = 0; r < 8; ++r) {
            int qrow = q0 + wave * 16 + rb + r;
            int d = n * 16 + nlane;
            o[(rowbase + qrow) * Dc + h * HDc + d] = f2bf(oacc[n][r] / lrow[r]);
        }
    }
}

// ---------------- lm head: logits = X(8x768,bf16) * wte^T ----------------
__global__ __launch_bounds__(256) void lmhead_kernel(
    const unsigned short* __restrict__ Xbf,   // [8, 768] bf16
    const float* __restrict__ wte,            // [V, D] f32
    float* __restrict__ out) {                // [8, V] f32
    __shared__ unsigned short Bs[256 * 32];    // [n][k] column-major tile
    int tid = threadIdx.x, lane = tid & 31, wave = tid >> 5;
    int n0 = blockIdx.x * 256;

    v8f acc[2];
    acc[0] = zero8(); acc[1] = zero8();

    for (int k0 = 0; k0 < Dc; k0 += 32) {
        {   // thread t owns vocab column n0+t: 32 k-values, contiguous in k
            int n = n0 + tid;
            union { uint4 q[4]; unsigned short u[32]; } tmp;
            if (n < Vc) {
                const float* src = &wte[(size_t)n * Dc + k0];
#pragma unroll
                for (int j = 0; j < 32; j += 4) {
                    float4 f = *(const float4*)(src + j);
                    tmp.u[j] = f2bf(f.x); tmp.u[j + 1] = f2bf(f.y);
                    tmp.u[j + 2] = f2bf(f.z); tmp.u[j + 3] = f2bf(f.w);
                }
            } else {
#pragma unroll
                for (int j = 0; j < 4; ++j) tmp.q[j] = make_uint4(0, 0, 0, 0);
            }
#pragma unroll
            for (int j = 0; j < 4; ++j)        // 4x b128 LDS stores
                *(uint4*)&Bs[tid * 32 + j * 8] = tmp.q[j];
        }
        __syncthreads();

        int m = lane & 15;
        v16bf a;
        if (m < Bc) {
            a = load_a_frag(Xbf, Dc, m, k0, lane);
        } else {
            union { v16bf v; uint4 q[2]; } z;
            z.q[0] = make_uint4(0, 0, 0, 0);
            z.q[1] = make_uint4(0, 0, 0, 0);
            a = z.v;
        }
#pragma unroll
        for (int n = 0; n < 2; ++n) {
            v16bf bfr = load_bT_frag(Bs, 32, wave * 32 + n * 16, 0, lane);
            acc[n] = __builtin_amdgcn_wmma_f32_16x16x32_bf16(
                false, a, false, bfr, (short)0, acc[n], false, false);
        }
        __syncthreads();
    }

    int nlane = lane & 15, rbv = (lane >> 4) * 8;
#pragma unroll
    for (int n = 0; n < 2; ++n) {
        int col = n0 + wave * 32 + n * 16 + nlane;
        if (col < Vc) {
#pragma unroll
            for (int r = 0; r < 8; ++r) {
                int row = rbv + r;
                if (row < Bc) out[(size_t)row * Vc + col] = acc[n][r];
            }
        }
    }
}

// ---------------- host orchestration ----------------
static inline size_t alignup(size_t x) { return (x + 255) & ~(size_t)255; }

extern "C" void kernel_launch(void* const* d_in, const int* in_sizes, int n_in,
                              void* d_out, int out_size, void* d_ws, size_t ws_size,
                              hipStream_t stream) {
    (void)in_sizes; (void)n_in; (void)out_size; (void)ws_size;

    const int*   idx   = (const int*)  d_in[0];
    const float* wte   = (const float*)d_in[1];
    const float* wpe   = (const float*)d_in[2];
    const float* ln1_w = (const float*)d_in[3];
    const float* ln1_b = (const float*)d_in[4];
    const float* qkv_w = (const float*)d_in[5];
    const float* qkv_b = (const float*)d_in[6];
    const float* po_w  = (const float*)d_in[7];
    const float* po_b  = (const float*)d_in[8];
    const float* ln2_w = (const float*)d_in[9];
    const float* ln2_b = (const float*)d_in[10];
    const float* fc_w  = (const float*)d_in[11];
    const float* fc_b  = (const float*)d_in[12];
    const float* pr_w  = (const float*)d_in[13];
    const float* pr_b  = (const float*)d_in[14];
    const float* lnf_w = (const float*)d_in[15];
    const float* lnf_b = (const float*)d_in[16];

    char* p = (char*)d_ws;
    float*          x      = (float*)p;          p += alignup((size_t)BTc * Dc * 4);
    unsigned short* hbuf   = (unsigned short*)p; p += alignup((size_t)BTc * Dc * 2);
    unsigned short* qkvbuf = (unsigned short*)p; p += alignup((size_t)BTc * 3 * Dc * 2);
    unsigned short* obuf   = (unsigned short*)p; p += alignup((size_t)BTc * Dc * 2);
    unsigned short* mlpbuf = (unsigned short*)p; p += alignup((size_t)BTc * 4 * Dc * 2);
    unsigned short* lnfbuf = (unsigned short*)p; p += alignup((size_t)Bc * Dc * 2);

    embed_kernel<<<BTc, 256, 0, stream>>>(idx, wte, wpe, x);

    for (int i = 0; i < Lc; ++i) {
        const float* qw = qkv_w + (size_t)i * Dc * 3 * Dc;
        const float* qb = qkv_b + (size_t)i * 3 * Dc;
        const float* pw = po_w  + (size_t)i * Dc * Dc;
        const float* pb = po_b  + (size_t)i * Dc;
        const float* fw = fc_w  + (size_t)i * Dc * 4 * Dc;
        const float* fb = fc_b  + (size_t)i * 4 * Dc;
        const float* rw = pr_w  + (size_t)i * 4 * Dc * Dc;
        const float* rb = pr_b  + (size_t)i * Dc;

        // attention
        ln_kernel<<<BTc, 256, 0, stream>>>(x, ln1_w + (size_t)i * Dc,
                                           ln1_b + (size_t)i * Dc, hbuf, 1, 0);
        gemm_bf16<0><<<dim3((3 * Dc) / 128, BTc / 64), 256, 0, stream>>>(
            hbuf, qw, qb, qkvbuf, nullptr, BTc, 3 * Dc, Dc);
        attn_kernel<<<dim3(Tc / 64, Hc, Bc), 128, 0, stream>>>(qkvbuf, obuf);
        gemm_bf16<2><<<dim3(Dc / 128, BTc / 64), 256, 0, stream>>>(
            obuf, pw, pb, nullptr, x, BTc, Dc, Dc);

        // MLP
        ln_kernel<<<BTc, 256, 0, stream>>>(x, ln2_w + (size_t)i * Dc,
                                           ln2_b + (size_t)i * Dc, hbuf, 1, 0);
        gemm_bf16<1><<<dim3((4 * Dc) / 128, BTc / 64), 256, 0, stream>>>(
            hbuf, fw, fb, mlpbuf, nullptr, BTc, 4 * Dc, Dc);
        gemm_bf16<2><<<dim3(Dc / 128, BTc / 64), 256, 0, stream>>>(
            mlpbuf, rw, rb, nullptr, x, BTc, Dc, 4 * Dc);
    }

    // final LN on the last token of each batch, then tied lm head
    ln_kernel<<<Bc, 256, 0, stream>>>(x, lnf_w, lnf_b, lnfbuf, Tc, Tc - 1);
    lmhead_kernel<<<(Vc + 255) / 256, 256, 0, stream>>>(lnfbuf, wte, (float*)d_out);
}
